// BehaviorGPTDecoder_68968584839855
// MI455X (gfx1250) — matrix operand order repaired and checked by reading, not compiled, not run on hardware
//
#include <hip/hip_runtime.h>
#include <math.h>

typedef __attribute__((ext_vector_type(16))) _Float16 v16h;
typedef __attribute__((ext_vector_type(8)))  float    v8f;
using f16 = _Float16;

#define A_  128
#define T_  64
#define N_  8192
#define M_  50000
#define H_  128
#define F_  64
#define TWO_PI_F 6.28318530717958647692f
#define PI_F     3.14159265358979323846f

// ---------------------------------------------------------------- device utils

__device__ __forceinline__ int imin_(int a, int b) { return a < b ? a : b; }

__device__ __forceinline__ float wrapang(float a) {
  float w = a + PI_F;
  w -= TWO_PI_F * floorf(w / TWO_PI_F);
  return w - PI_F;
}

__device__ __forceinline__ void atomicMaxFloat(float* addr, float val) {
  if (val >= 0.f) atomicMax((int*)addr, __float_as_int(val));
  else            atomicMin((unsigned int*)addr, __float_as_uint(val));
}

// A fragment (16x32 f16, CDNA5 layout) gathered from a row-major f16 LDS tile.
__device__ __forceinline__ v16h load_a_frag(const f16* tile, int ldm, int k0, int lane) {
  const int hg = lane >> 4;
  const f16* row = tile + (size_t)(lane & 15) * ldm + k0;
  v16h a;
#pragma unroll
  for (int e = 0; e < 16; ++e) {
    int k = e + ((e >= 8) ? 8 : 0) + hg * 8;
    a[e] = row[k];
  }
  return a;
}

__device__ __forceinline__ v8f wmma16(v16h a, v16h b, v8f c) {
  return __builtin_amdgcn_wmma_f32_16x16x32_f16(false, a, false, b, (short)0, c, false, false);
}

// B fragment from pre-packed global weights (512 halfs per frag, lane-major).
__device__ __forceinline__ v16h load_b_frag(const f16* packed, int frag) {
  return *(const v16h*)(packed + (size_t)frag * 512 + (size_t)(threadIdx.x & 31) * 16);
}

// C = A(16x128, f16 LDS) x W(128x128 packed) + bias -> f32 LDS tile (16x128)
__device__ __forceinline__ void gemm128_to_lds(const f16* a_tile, const f16* Wp,
                                               const float* bias, float* out, int lane) {
  const int col16 = lane & 15, rb = (lane >> 4) * 8;
  for (int nt = 0; nt < 8; ++nt) {
    v8f c = {};
#pragma unroll
    for (int kc = 0; kc < 4; ++kc)
      c = wmma16(load_a_frag(a_tile, 128, kc * 32, lane), load_b_frag(Wp, nt * 4 + kc), c);
    int col = nt * 16 + col16;
    float bv = bias ? bias[col] : 0.f;
#pragma unroll
    for (int r = 0; r < 8; ++r) out[(rb + r) * 128 + col] = c[r] + bv;
  }
}

// Same, but scatter result rows to global (f16 and/or f32), clamped rows.
__device__ __forceinline__ void gemm128_to_global(const f16* a_tile, const f16* Wp,
                                                  const float* bias, f16* out16, float* outf,
                                                  int row0, int n, int lane) {
  const int col16 = lane & 15, rb = (lane >> 4) * 8;
  for (int nt = 0; nt < 8; ++nt) {
    v8f c = {};
#pragma unroll
    for (int kc = 0; kc < 4; ++kc)
      c = wmma16(load_a_frag(a_tile, 128, kc * 32, lane), load_b_frag(Wp, nt * 4 + kc), c);
    int col = nt * 16 + col16;
    float bv = bias ? bias[col] : 0.f;
#pragma unroll
    for (int r = 0; r < 8; ++r) {
      int row = imin_(row0 + rb + r, n - 1);
      float v = c[r] + bv;
      if (out16) out16[(size_t)row * H_ + col] = (f16)v;
      if (outf)  outf [(size_t)row * H_ + col] = v;
    }
  }
}

// LayerNorm rows of a 16x128 f32 LDS tile -> f16 LDS tile (lanes 0..15, one row each).
__device__ __forceinline__ void ln_rows_to_f16(const float* src, f16* dst, const float* g,
                                               const float* b, int lane, bool relu_) {
  if (lane < 16) {
    const float* row = src + lane * 128;
    float mu = 0.f, m2 = 0.f;
    for (int i = 0; i < 128; ++i) { float v = row[i]; mu += v; m2 += v * v; }
    mu *= (1.f / 128.f);
    float var = m2 * (1.f / 128.f) - mu * mu;
    float rs = rsqrtf(fmaxf(var, 0.f) + 1e-5f);
    f16* o = dst + lane * 128;
    for (int i = 0; i < 128; ++i) {
      float v = (row[i] - mu) * rs * g[i] + b[i];
      if (relu_) v = fmaxf(v, 0.f);
      o[i] = (f16)v;
    }
  }
}

// ---------------------------------------------------------------- weight packing

// Pack f32 KxN weight into fragment-contiguous f16: frag = nt*KC+kc, 512 halfs/frag,
// B layout: element e of lane -> K = kc*32 + e + (e>=8?8:0) + 8*(lane>>4), col = nt*16 + (lane&15).
__global__ void pack_w16_kernel(const float* W, f16* dst, int K, int Ncols, int KC, int NT) {
  int t = blockIdx.x * blockDim.x + threadIdx.x;
  int total = KC * NT * 32;
  if (t >= total) return;
  int lane = t & 31, frag = t >> 5;
  int kc = frag % KC, nt = frag / KC;
  f16* o = dst + (size_t)(nt * KC + kc) * 512 + (size_t)lane * 16;
  int col = nt * 16 + (lane & 15);
  int kb = kc * 32 + 8 * (lane >> 4);
  for (int e = 0; e < 16; ++e) {
    int k = kb + e + ((e >= 8) ? 8 : 0);
    o[e] = (f16)((k < K) ? W[(size_t)k * Ncols + col] : 0.f);
  }
}

// ---------------------------------------------------------------- feature kernels

__global__ void xa_feat_kernel(const float* head_a, const float* vel, const float* len,
                               const float* wid, const float* hei, float* out) {
  int n = blockIdx.x * blockDim.x + threadIdx.x;
  if (n >= N_) return;
  float sh, ch; __sincosf(head_a[n], &sh, &ch);
  float vx = vel[n * 2], vy = vel[n * 2 + 1];
  out[n * 5 + 0] = sqrtf(vx * vx + vy * vy);
  out[n * 5 + 1] = atan2f(ch * vy - sh * vx, ch * vx + sh * vy);
  out[n * 5 + 2] = len[n];
  out[n * 5 + 3] = wid[n];
  out[n * 5 + 4] = hei[n];
}

__global__ void catidx_kernel(const int* agent_type, int* idx) {
  int n = blockIdx.x * blockDim.x + threadIdx.x;
  if (n < N_) idx[n] = agent_type[n / T_];
}

__global__ void rel_t_kernel(const int* ei, int E, const float* pos, const float* head, float* out) {
  int e = blockIdx.x * blockDim.x + threadIdx.x;
  if (e >= E) return;
  int s = ei[e], d = ei[E + e];
  float rx = pos[s * 2] - pos[d * 2], ry = pos[s * 2 + 1] - pos[d * 2 + 1];
  float sh, ch; __sincosf(head[d], &sh, &ch);
  out[e * 4 + 0] = sqrtf(rx * rx + ry * ry);
  out[e * 4 + 1] = atan2f(ch * ry - sh * rx, ch * rx + sh * ry);
  out[e * 4 + 2] = wrapang(head[s] - head[d]);
  out[e * 4 + 3] = (float)(s - d);
}

__global__ void rel_m2a_kernel(const int* ei, int E, const float* pos_m, const float* orient_m,
                               const float* pos, const float* head, float* out) {
  int e = blockIdx.x * blockDim.x + threadIdx.x;
  if (e >= E) return;
  int s = ei[e], d = ei[E + e];
  float rx = pos_m[s * 2] - pos[d * 2], ry = pos_m[s * 2 + 1] - pos[d * 2 + 1];
  float sh, ch; __sincosf(head[d], &sh, &ch);
  out[e * 3 + 0] = sqrtf(rx * rx + ry * ry);
  out[e * 3 + 1] = atan2f(ch * ry - sh * rx, ch * rx + sh * ry);
  out[e * 3 + 2] = wrapang(orient_m[s] - head[d]);
}

__global__ void rel_a2a_kernel(const int* ei, int E, const float* pos_a, const float* head_a,
                               float* out) {
  int e = blockIdx.x * blockDim.x + threadIdx.x;
  if (e >= E) return;
  int s = ei[e], d = ei[E + e];
  int sn = (s % A_) * T_ + (s / A_);
  int dn = (d % A_) * T_ + (d / A_);
  float rx = pos_a[sn * 2] - pos_a[dn * 2], ry = pos_a[sn * 2 + 1] - pos_a[dn * 2 + 1];
  float hd = head_a[dn];
  float sh, ch; __sincosf(hd, &sh, &ch);
  out[e * 3 + 0] = sqrtf(rx * rx + ry * ry);
  out[e * 3 + 1] = atan2f(ch * ry - sh * rx, ch * rx + sh * ry);
  out[e * 3 + 2] = wrapang(head_a[sn] - hd);
}

__global__ void fillf_kernel(float* p, float v, long n) {
  long t = (long)blockIdx.x * blockDim.x + threadIdx.x;
  if (t < n) p[t] = v;
}

__global__ void copyf_kernel(const float* s, float* d, long n) {
  long t = (long)blockIdx.x * blockDim.x + threadIdx.x;
  if (t < n) d[t] = s[t];
}

// fwd: x_perm[t*A+a] = x[a*T+t]; bwd: x[a*T+t] = x_perm[t*A+a]
__global__ void permute_kernel(const float* src, float* dst, int fwd) {
  long t = (long)blockIdx.x * blockDim.x + threadIdx.x;
  if (t >= (long)N_ * H_) return;
  int c = (int)(t & 127), n = (int)(t >> 7);
  if (fwd) { int a = n / T_, tt = n - a * T_; dst[((size_t)(tt * A_ + a) << 7) + c] = src[t]; }
  else     { int tt = n / A_, a = n - tt * A_; dst[((size_t)(a * T_ + tt) << 7) + c] = src[t]; }
}

// ---------------------------------------------------------------- fourier embedding

struct FourierArgs {
  const float* x; int n, din;
  const float* freqs;
  const f16* W1p; const float* b1; const float* g1; const float* bt1;
  const f16* W2p; const float* b2;
  const float* g_out; const float* b_out;
  const f16* Woutp; const float* bout;
  const float* cat_table; const int* cat_idx;
  const unsigned char* valid;
  float* outf; f16* outh;
};

__global__ void __launch_bounds__(128) fourier_kernel(FourierArgs Ag) {
  __shared__ f16   s_a[4][16 * 160];
  __shared__ float s_h[4][16 * 128];
  int w = threadIdx.x >> 5, lane = threadIdx.x & 31;
  int row0 = (blockIdx.x * 4 + w) * 16;
  const int col16 = lane & 15, rb = (lane >> 4) * 8;
  v8f acc[8] = {};
  for (int d = 0; d < Ag.din; ++d) {
    if (lane < 16) {
      int r = imin_(row0 + lane, Ag.n - 1);
      float xv = Ag.x[(size_t)r * Ag.din + d];
      f16* fr = s_a[w] + lane * 160;
      const float* fq = Ag.freqs + d * F_;
      for (int f = 0; f < F_; ++f) {
        float sv, cv; __sincosf(xv * fq[f] * TWO_PI_F, &sv, &cv);
        fr[f] = (f16)cv; fr[F_ + f] = (f16)sv;
      }
      fr[128] = (f16)xv;
      for (int k2 = 129; k2 < 160; ++k2) fr[k2] = (f16)0.f;
    }
    __syncthreads();
    const f16* W1 = Ag.W1p + (size_t)d * (5 * 8 * 512);
    for (int nt = 0; nt < 8; ++nt) {
      v8f c = {};
#pragma unroll
      for (int kc = 0; kc < 5; ++kc)
        c = wmma16(load_a_frag(s_a[w], 160, kc * 32, lane), load_b_frag(W1, nt * 5 + kc), c);
      int col = nt * 16 + col16;
      float bv = Ag.b1[d * H_ + col];
#pragma unroll
      for (int r = 0; r < 8; ++r) s_h[w][(rb + r) * 128 + col] = c[r] + bv;
    }
    __syncthreads();
    ln_rows_to_f16(s_h[w], s_a[w], Ag.g1 + d * H_, Ag.bt1 + d * H_, lane, true);
    __syncthreads();
    const f16* W2 = Ag.W2p + (size_t)d * (4 * 8 * 512);
    for (int nt = 0; nt < 8; ++nt) {
      v8f c = acc[nt];
#pragma unroll
      for (int kc = 0; kc < 4; ++kc)
        c = wmma16(load_a_frag(s_a[w], 128, kc * 32, lane), load_b_frag(W2, nt * 4 + kc), c);
      int col = nt * 16 + col16;
      float bv = Ag.b2[d * H_ + col];
#pragma unroll
      for (int r = 0; r < 8; ++r) c[r] += bv;
      acc[nt] = c;
    }
    __syncthreads();
  }
  if (Ag.cat_idx) {
    for (int nt = 0; nt < 8; ++nt) {
      int col = nt * 16 + col16;
      for (int r = 0; r < 8; ++r) {
        int row = imin_(row0 + rb + r, Ag.n - 1);
        acc[nt][r] += Ag.cat_table[(size_t)Ag.cat_idx[row] * H_ + col];
      }
    }
  }
  for (int nt = 0; nt < 8; ++nt) {
    int col = nt * 16 + col16;
    for (int r = 0; r < 8; ++r) s_h[w][(rb + r) * 128 + col] = acc[nt][r];
  }
  __syncthreads();
  ln_rows_to_f16(s_h[w], s_a[w], Ag.g_out, Ag.b_out, lane, true);
  __syncthreads();
  for (int nt = 0; nt < 8; ++nt) {
    v8f c = {};
#pragma unroll
    for (int kc = 0; kc < 4; ++kc)
      c = wmma16(load_a_frag(s_a[w], 128, kc * 32, lane), load_b_frag(Ag.Woutp, nt * 4 + kc), c);
    int col = nt * 16 + col16;
    float bv = Ag.bout[col];
    for (int r = 0; r < 8; ++r) {
      int row = imin_(row0 + rb + r, Ag.n - 1);
      float v = c[r] + bv;
      if (Ag.valid && !Ag.valid[row]) v = 0.f;
      if (Ag.outf) Ag.outf[(size_t)row * H_ + col] = v;
      if (Ag.outh) Ag.outh[(size_t)row * H_ + col] = (f16)v;
    }
  }
}

// ---------------------------------------------------------------- attention: pre

struct PreArgs {
  const float* x; int n;
  const float *g_src, *b_src, *g_dst, *b_dst;
  const f16 *Wqp, *Wkp, *Wvp; const float* bq;
  f16 *xs16, *xd16, *ks16, *vs16; float* q;
  int do_src, do_dst, share;
};

__global__ void __launch_bounds__(128) attn_pre_kernel(PreArgs Ag) {
  __shared__ float sx[4][2048];
  __shared__ f16   sa[4][2048];
  int w = threadIdx.x >> 5, lane = threadIdx.x & 31;
  int row0 = (blockIdx.x * 4 + w) * 16;
  if (lane < 16) {
    int r = imin_(row0 + lane, Ag.n - 1);
    const float* src = Ag.x + (size_t)r * H_;
    float* d = sx[w] + lane * 128;
    for (int i = 0; i < 128; ++i) d[i] = src[i];
  }
  __syncthreads();
  if (Ag.do_src) {
    ln_rows_to_f16(sx[w], sa[w], Ag.g_src, Ag.b_src, lane, false);
    __syncthreads();
    if (lane < 16) {
      int r = imin_(row0 + lane, Ag.n - 1);
      f16* o = Ag.xs16 + (size_t)r * H_;
      const f16* s = sa[w] + lane * 128;
      for (int i = 0; i < 128; ++i) o[i] = s[i];
    }
    gemm128_to_global(sa[w], Ag.Wkp, nullptr, Ag.ks16, nullptr, row0, Ag.n, lane);
    gemm128_to_global(sa[w], Ag.Wvp, nullptr, Ag.vs16, nullptr, row0, Ag.n, lane);
  }
  if (Ag.do_dst) {
    if (!Ag.share) {
      __syncthreads();
      ln_rows_to_f16(sx[w], sa[w], Ag.g_dst, Ag.b_dst, lane, false);
      __syncthreads();
      if (lane < 16) {
        int r = imin_(row0 + lane, Ag.n - 1);
        f16* o = Ag.xd16 + (size_t)r * H_;
        const f16* s = sa[w] + lane * 128;
        for (int i = 0; i < 128; ++i) o[i] = s[i];
      }
    }
    gemm128_to_global(sa[w], Ag.Wqp, Ag.bq, nullptr, Ag.q, row0, Ag.n, lane);
  }
}

// ---------------------------------------------------------------- attention: edge k/v + sim

struct EdgeArgs {
  const f16* r16; int E;
  const float *g_r, *b_r;
  const f16 *Wkrp, *Wvrp;
  const f16 *ks16, *vs16; const float* q;
  const int *src, *dst;
  float* sim; f16* vedge;
};

__global__ void __launch_bounds__(128) edge_kv_kernel(EdgeArgs Ag) {
  __shared__ f16   sa[4][2048];
  __shared__ float sh[4][2048];
  int w = threadIdx.x >> 5, lane = threadIdx.x & 31;
  int e0 = (blockIdx.x * 4 + w) * 16;
  if (lane == 0 && e0 + 16 < Ag.E)
    __builtin_prefetch(Ag.r16 + (size_t)(e0 + 16) * H_, 0, 1);
  if (lane < 16) {
    int e = imin_(e0 + lane, Ag.E - 1);
    const f16* rr = Ag.r16 + (size_t)e * H_;
    float mu = 0.f, m2 = 0.f;
    for (int i = 0; i < 128; ++i) { float v = (float)rr[i]; mu += v; m2 += v * v; }
    mu *= (1.f / 128.f);
    float var = m2 * (1.f / 128.f) - mu * mu;
    float rs = rsqrtf(fmaxf(var, 0.f) + 1e-5f);
    f16* o = sa[w] + lane * 128;
    for (int i = 0; i < 128; ++i)
      o[i] = (f16)(((float)rr[i] - mu) * rs * Ag.g_r[i] + Ag.b_r[i]);
  }
  __syncthreads();
  gemm128_to_lds(sa[w], Ag.Wkrp, nullptr, sh[w], lane);
  __syncthreads();
  if (lane < 16) {
    int e = imin_(e0 + lane, Ag.E - 1);
    int s = Ag.src[e], d = Ag.dst[e];
    const f16* ks = Ag.ks16 + (size_t)s * H_;
    const float* qr = Ag.q + (size_t)d * H_;
    const float* kr = sh[w] + lane * 128;
    float* so = Ag.sim + (size_t)e * 8;
    for (int h = 0; h < 8; ++h) {
      float a = 0.f;
      for (int j = 0; j < 16; ++j) {
        int i = h * 16 + j;
        a += (kr[i] + (float)ks[i]) * qr[i];
      }
      so[h] = a * 0.25f;
    }
  }
  __syncthreads();
  gemm128_to_lds(sa[w], Ag.Wvrp, nullptr, sh[w], lane);
  __syncthreads();
  if (lane < 16) {
    int e = imin_(e0 + lane, Ag.E - 1);
    int s = Ag.src[e];
    const f16* vs = Ag.vs16 + (size_t)s * H_;
    const float* vr = sh[w] + lane * 128;
    f16* vo = Ag.vedge + (size_t)e * H_;
    for (int i = 0; i < 128; ++i) vo[i] = (f16)(vr[i] + (float)vs[i]);
  }
}

// ---------------------------------------------------------------- segment softmax

__global__ void seg_max_kernel(const float* sim, const int* dst, int E, float* m) {
  long t = (long)blockIdx.x * blockDim.x + threadIdx.x;
  if (t >= (long)E * 8) return;
  int e = (int)(t >> 3), h = (int)(t & 7);
  atomicMaxFloat(&m[(size_t)dst[e] * 8 + h], sim[t]);
}

__global__ void seg_expsum_kernel(float* sim, const int* dst, int E, const float* m, float* den) {
  long t = (long)blockIdx.x * blockDim.x + threadIdx.x;
  if (t >= (long)E * 8) return;
  int e = (int)(t >> 3), h = (int)(t & 7);
  float w = __expf(sim[t] - m[(size_t)dst[e] * 8 + h]);
  sim[t] = w;
  atomicAdd(&den[(size_t)dst[e] * 8 + h], w);
}

__global__ void seg_agg_kernel(const float* wgt, const float* den, const f16* vedge,
                               const int* dst, int E, float* agg) {
  long t = (long)blockIdx.x * blockDim.x + threadIdx.x;
  if (t >= (long)E * H_) return;
  int e = (int)(t >> 7), c = (int)(t & 127), h = c >> 4;
  int d = dst[e];
  float a = wgt[(size_t)e * 8 + h] / den[(size_t)d * 8 + h];
  atomicAdd(&agg[(size_t)d * H_ + c], a * (float)vedge[t]);
}

// ---------------------------------------------------------------- attention: post (gate + out + FFN)

struct PostArgs {
  float* x; // residual in/out (per-row independent -> in-place safe)
  const f16* xd16; const float* agg; int n;
  const f16 *Wgp, *Wsp, *Wop, *W1fp, *W2fp;
  const float *bg, *bs, *bo, *g_ff, *b_ff, *b1f, *b2f;
};

__global__ void __launch_bounds__(64) attn_post_kernel(PostArgs Ag) {
  __shared__ f16   s16a[2][2048];
  __shared__ f16   s16b[2][2048];
  __shared__ float sf1[2][2048];
  __shared__ float sf2[2][2048];
  int w = threadIdx.x >> 5, lane = threadIdx.x & 31;
  int row0 = (blockIdx.x * 2 + w) * 16;
  const int col16 = lane & 15, rb = (lane >> 4) * 8;
  if (lane < 16) {
    int r = imin_(row0 + lane, Ag.n - 1);
    const float* ar = Ag.agg + (size_t)r * H_;
    const f16* xr = Ag.xd16 + (size_t)r * H_;
    for (int i = 0; i < 128; ++i) {
      float v = ar[i];
      sf2[w][lane * 128 + i] = v;
      s16a[w][lane * 128 + i] = (f16)v;
      s16b[w][lane * 128 + i] = xr[i];
    }
  }
  __syncthreads();
  // gate = sigmoid([agg | xd] @ Wg + bg)
  for (int nt = 0; nt < 8; ++nt) {
    v8f c = {};
#pragma unroll
    for (int kc = 0; kc < 8; ++kc) {
      const f16* at = (kc < 4) ? s16a[w] : s16b[w];
      c = wmma16(load_a_frag(at, 128, (kc & 3) * 32, lane), load_b_frag(Ag.Wgp, nt * 8 + kc), c);
    }
    int col = nt * 16 + col16;
    float bv = Ag.bg[col];
#pragma unroll
    for (int r = 0; r < 8; ++r)
      sf1[w][(rb + r) * 128 + col] = 1.f / (1.f + __expf(-(c[r] + bv)));
  }
  __syncthreads();
  // inner = g*(xd@Ws+bs) + (1-g)*agg
  for (int nt = 0; nt < 8; ++nt) {
    v8f c = {};
#pragma unroll
    for (int kc = 0; kc < 4; ++kc)
      c = wmma16(load_a_frag(s16b[w], 128, kc * 32, lane), load_b_frag(Ag.Wsp, nt * 4 + kc), c);
    int col = nt * 16 + col16;
    float bv = Ag.bs[col];
#pragma unroll
    for (int r = 0; r < 8; ++r) {
      int idx = (rb + r) * 128 + col;
      float gv = sf1[w][idx], ag = sf2[w][idx];
      s16a[w][idx] = (f16)(gv * (c[r] + bv) + (1.f - gv) * ag);
    }
  }
  __syncthreads();
  // x_new = inner @ Wo + bo + x
  for (int nt = 0; nt < 8; ++nt) {
    v8f c = {};
#pragma unroll
    for (int kc = 0; kc < 4; ++kc)
      c = wmma16(load_a_frag(s16a[w], 128, kc * 32, lane), load_b_frag(Ag.Wop, nt * 4 + kc), c);
    int col = nt * 16 + col16;
    float bv = Ag.bo[col];
#pragma unroll
    for (int r = 0; r < 8; ++r) {
      int row = imin_(row0 + rb + r, Ag.n - 1);
      sf2[w][(rb + r) * 128 + col] = c[r] + bv + Ag.x[(size_t)row * H_ + col];
    }
  }
  __syncthreads();
  ln_rows_to_f16(sf2[w], s16b[w], Ag.g_ff, Ag.b_ff, lane, false);
  __syncthreads();
  // FFN: x_new + relu(h@W1f+b1f)@W2f + b2f  (512-wide in 4 chunks of 128)
  v8f acc2[8] = {};
  for (int cc = 0; cc < 4; ++cc) {
    for (int nt2 = 0; nt2 < 8; ++nt2) {
      v8f c = {};
#pragma unroll
      for (int kc = 0; kc < 4; ++kc)
        c = wmma16(load_a_frag(s16b[w], 128, kc * 32, lane),
                   load_b_frag(Ag.W1fp, (cc * 8 + nt2) * 4 + kc), c);
      int col = nt2 * 16 + col16;
      float bv = Ag.b1f[cc * 128 + col];
#pragma unroll
      for (int r = 0; r < 8; ++r)
        s16a[w][(rb + r) * 128 + col] = (f16)fmaxf(c[r] + bv, 0.f);
    }
    __syncthreads();
    for (int nt = 0; nt < 8; ++nt) {
      v8f c = acc2[nt];
#pragma unroll
      for (int kc = 0; kc < 4; ++kc)
        c = wmma16(load_a_frag(s16a[w], 128, kc * 32, lane),
                   load_b_frag(Ag.W2fp, nt * 16 + cc * 4 + kc), c);
      acc2[nt] = c;
    }
    __syncthreads();
  }
  for (int nt = 0; nt < 8; ++nt) {
    int col = nt * 16 + col16;
    for (int r = 0; r < 8; ++r) {
      int row = imin_(row0 + rb + r, Ag.n - 1);
      Ag.x[(size_t)row * H_ + col] = sf2[w][(rb + r) * 128 + col] + acc2[nt][r] + Ag.b2f[col];
    }
  }
}

// ---------------------------------------------------------------- host side

struct FourierW { const float *W1, *W2, *Wout, *b1, *b2, *b_out, *bout, *bt1, *freqs, *g1, *g_out; };
struct AttnW {
  const float *W1f, *W2f, *Wg, *Wk, *Wkr, *Wo, *Wq, *Ws, *Wv, *Wvr,
              *b1f, *b2f, *b_dst, *b_ff, *b_r, *b_src, *bg, *bo, *bq, *bs,
              *g_dst, *g_ff, *g_r, *g_src;
};
struct FourierPk { f16 *W1, *W2, *Wout; };
struct AttnPk { f16 *Wq, *Wk, *Wv, *Wkr, *Wvr, *Ws, *Wg, *Wo, *W1f, *W2f; };

static void takeF(FourierW& f, void* const* din, int& i) {
  const float** p = (const float**)&f;
  for (int k = 0; k < 11; ++k) p[k] = (const float*)din[i++];
}
static void takeA(AttnW& a, void* const* din, int& i) {
  const float** p = (const float**)&a;
  for (int k = 0; k < 24; ++k) p[k] = (const float*)din[i++];
}

extern "C" void kernel_launch(void* const* d_in, const int* in_sizes, int n_in,
                              void* d_out, int out_size, void* d_ws, size_t ws_size,
                              hipStream_t stream) {
  (void)n_in; (void)out_size; (void)ws_size;
  const float* pos_a       = (const float*)d_in[0];
  const float* head_a      = (const float*)d_in[1];
  const float* vel         = (const float*)d_in[2];
  const float* len         = (const float*)d_in[3];
  const float* wid         = (const float*)d_in[4];
  const float* hei         = (const float*)d_in[5];
  const unsigned char* mask = (const unsigned char*)d_in[6];
  const int* agent_type    = (const int*)d_in[7];
  const float* pos_m       = (const float*)d_in[8];
  const float* orient_m    = (const float*)d_in[9];
  const float* magnitude_m = (const float*)d_in[10];
  const int* map_type      = (const int*)d_in[11];
  const int* ei_patch = (const int*)d_in[12]; int Ep = in_sizes[12] / 2;
  const int* ei_t     = (const int*)d_in[13]; int Et = in_sizes[13] / 2;
  const int* ei_m2a   = (const int*)d_in[14]; int Em = in_sizes[14] / 2;
  const int* ei_a2a   = (const int*)d_in[15]; int Ea = in_sizes[15] / 2;

  // params: jax pytree flatten, dict keys sorted
  int i = 16;
  AttnW a2aW[2]; takeA(a2aW[0], d_in, i); takeA(a2aW[1], d_in, i);
  AttnW m2aW[2]; takeA(m2aW[0], d_in, i); takeA(m2aW[1], d_in, i);
  FourierW r_a2aW; takeF(r_a2aW, d_in, i);
  FourierW r_m2aW; takeF(r_m2aW, d_in, i);
  FourierW r_patchW; takeF(r_patchW, d_in, i);
  FourierW r_tW; takeF(r_tW, d_in, i);
  AttnW tW[2]; takeA(tW[0], d_in, i); takeA(tW[1], d_in, i);
  AttnW topW; takeA(topW, d_in, i);
  const float* type_a_emb = (const float*)d_in[i++];
  const float* type_m_emb = (const float*)d_in[i++];
  FourierW x_aW; takeF(x_aW, d_in, i);
  FourierW x_mW; takeF(x_mW, d_in, i);

  size_t off = 0;
  auto alloc = [&](size_t bytes) -> void* {
    off = (off + 255) & ~(size_t)255;
    void* p = (char*)d_ws + off;
    off += bytes;
    return p;
  };
  auto packmat = [&](const float* W, int K, int Nc) -> f16* {
    int KC = (K + 31) / 32, NT = Nc / 16;
    f16* dst = (f16*)alloc((size_t)KC * NT * 512 * sizeof(f16));
    int total = KC * NT * 32;
    pack_w16_kernel<<<(total + 255) / 256, 256, 0, stream>>>(W, dst, K, Nc, KC, NT);
    return dst;
  };
  auto packattn = [&](const AttnW& w) -> AttnPk {
    AttnPk p;
    p.Wq = packmat(w.Wq, 128, 128);  p.Wk = packmat(w.Wk, 128, 128);
    p.Wv = packmat(w.Wv, 128, 128);  p.Wkr = packmat(w.Wkr, 128, 128);
    p.Wvr = packmat(w.Wvr, 128, 128); p.Ws = packmat(w.Ws, 128, 128);
    p.Wg = packmat(w.Wg, 256, 128);  p.Wo = packmat(w.Wo, 128, 128);
    p.W1f = packmat(w.W1f, 128, 512); p.W2f = packmat(w.W2f, 512, 128);
    return p;
  };
  auto packfour = [&](const FourierW& w, int din) -> FourierPk {
    FourierPk p;
    p.W1 = (f16*)alloc((size_t)din * 5 * 8 * 512 * sizeof(f16));
    for (int d = 0; d < din; ++d)
      pack_w16_kernel<<<(5 * 8 * 32 + 255) / 256, 256, 0, stream>>>(
          w.W1 + (size_t)d * 129 * 128, p.W1 + (size_t)d * 5 * 8 * 512, 129, 128, 5, 8);
    p.W2 = (f16*)alloc((size_t)din * 4 * 8 * 512 * sizeof(f16));
    for (int d = 0; d < din; ++d)
      pack_w16_kernel<<<(4 * 8 * 32 + 255) / 256, 256, 0, stream>>>(
          w.W2 + (size_t)d * 128 * 128, p.W2 + (size_t)d * 4 * 8 * 512, 128, 128, 4, 8);
    p.Wout = packmat(w.Wout, 128, 128);
    return p;
  };

  FourierPk pk_xa = packfour(x_aW, 5), pk_xm = packfour(x_mW, 1);
  FourierPk pk_rp = packfour(r_patchW, 4), pk_rt = packfour(r_tW, 4);
  FourierPk pk_rm = packfour(r_m2aW, 3), pk_ra = packfour(r_a2aW, 3);
  AttnPk pk_top = packattn(topW);
  AttnPk pk_t[2] = { packattn(tW[0]), packattn(tW[1]) };
  AttnPk pk_m[2] = { packattn(m2aW[0]), packattn(m2aW[1]) };
  AttnPk pk_a[2] = { packattn(a2aW[0]), packattn(a2aW[1]) };

  int Emax = Ep; if (Et > Emax) Emax = Et; if (Em > Emax) Emax = Em; if (Ea > Emax) Emax = Ea;

  float* xa_cont  = (float*)alloc((size_t)N_ * 5 * 4);
  int*   cat_idx  = (int*)alloc((size_t)N_ * 4);
  float* rel_p    = (float*)alloc((size_t)Ep * 4 * 4);
  float* rel_tt   = (float*)alloc((size_t)Et * 4 * 4);
  float* rel_m    = (float*)alloc((size_t)Em * 3 * 4);
  float* rel_a    = (float*)alloc((size_t)Ea * 3 * 4);
  float* x_t      = (float*)alloc((size_t)N_ * H_ * 4);
  float* x_m      = (float*)alloc((size_t)M_ * H_ * 4);
  float* x_perm   = (float*)alloc((size_t)N_ * H_ * 4);
  f16* r16_patch  = (f16*)alloc((size_t)Ep * H_ * 2);
  f16* r16_t      = (f16*)alloc((size_t)Et * H_ * 2);
  f16* r16_m2a    = (f16*)alloc((size_t)Em * H_ * 2);
  f16* r16_a2a    = (f16*)alloc((size_t)Ea * H_ * 2);
  f16* xs16       = (f16*)alloc((size_t)M_ * H_ * 2);
  f16* xd16       = (f16*)alloc((size_t)N_ * H_ * 2);
  f16* ks16       = (f16*)alloc((size_t)M_ * H_ * 2);
  f16* vs16       = (f16*)alloc((size_t)M_ * H_ * 2);
  float* qbuf     = (float*)alloc((size_t)N_ * H_ * 4);
  float* simb     = (float*)alloc((size_t)Emax * 8 * 4);
  float* mbuf     = (float*)alloc((size_t)N_ * 8 * 4);
  float* denb     = (float*)alloc((size_t)N_ * 8 * 4);
  f16* vedge      = (f16*)alloc((size_t)Emax * H_ * 2);
  float* aggb     = (float*)alloc((size_t)N_ * H_ * 4);

  // features
  xa_feat_kernel<<<(N_ + 255) / 256, 256, 0, stream>>>(head_a, vel, len, wid, hei, xa_cont);
  catidx_kernel<<<(N_ + 255) / 256, 256, 0, stream>>>(agent_type, cat_idx);
  rel_t_kernel<<<(Ep + 255) / 256, 256, 0, stream>>>(ei_patch, Ep, pos_a, head_a, rel_p);
  rel_t_kernel<<<(Et + 255) / 256, 256, 0, stream>>>(ei_t, Et, pos_a, head_a, rel_tt);
  rel_m2a_kernel<<<(Em + 255) / 256, 256, 0, stream>>>(ei_m2a, Em, pos_m, orient_m, pos_a, head_a, rel_m);
  rel_a2a_kernel<<<(Ea + 255) / 256, 256, 0, stream>>>(ei_a2a, Ea, pos_a, head_a, rel_a);

  auto runF = [&](const FourierW& w, const FourierPk& pk, int din, const float* x, int n,
                  const float* cat_table, const int* cidx, const unsigned char* valid,
                  float* outf, f16* outh) {
    FourierArgs Ag{};
    Ag.x = x; Ag.n = n; Ag.din = din; Ag.freqs = w.freqs;
    Ag.W1p = pk.W1; Ag.b1 = w.b1; Ag.g1 = w.g1; Ag.bt1 = w.bt1;
    Ag.W2p = pk.W2; Ag.b2 = w.b2;
    Ag.g_out = w.g_out; Ag.b_out = w.b_out; Ag.Woutp = pk.Wout; Ag.bout = w.bout;
    Ag.cat_table = cat_table; Ag.cat_idx = cidx; Ag.valid = valid;
    Ag.outf = outf; Ag.outh = outh;
    int tiles = (n + 15) / 16;
    fourier_kernel<<<(tiles + 3) / 4, 128, 0, stream>>>(Ag);
  };

  runF(x_aW, pk_xa, 5, xa_cont, N_, type_a_emb, cat_idx, mask, x_t, nullptr);
  runF(x_mW, pk_xm, 1, magnitude_m, M_, type_m_emb, map_type, nullptr, x_m, nullptr);
  runF(r_patchW, pk_rp, 4, rel_p, Ep, nullptr, nullptr, nullptr, nullptr, r16_patch);
  runF(r_tW, pk_rt, 4, rel_tt, Et, nullptr, nullptr, nullptr, nullptr, r16_t);
  runF(r_m2aW, pk_rm, 3, rel_m, Em, nullptr, nullptr, nullptr, nullptr, r16_m2a);
  runF(r_a2aW, pk_ra, 3, rel_a, Ea, nullptr, nullptr, nullptr, nullptr, r16_a2a);

  auto runAttn = [&](const AttnW& w, const AttnPk& pk, float* xsrc, float* xdst, int nsrc,
                     const f16* r16, const int* ei, int E, bool bip) {
    if (!bip) {
      PreArgs P{};
      P.x = xdst; P.n = N_;
      P.g_src = w.g_src; P.b_src = w.b_src;
      P.Wqp = pk.Wq; P.Wkp = pk.Wk; P.Wvp = pk.Wv; P.bq = w.bq;
      P.xs16 = xs16; P.xd16 = xs16; P.ks16 = ks16; P.vs16 = vs16; P.q = qbuf;
      P.do_src = 1; P.do_dst = 1; P.share = 1;
      int tiles = (N_ + 15) / 16;
      attn_pre_kernel<<<(tiles + 3) / 4, 128, 0, stream>>>(P);
    } else {
      PreArgs Ps{};
      Ps.x = xsrc; Ps.n = nsrc;
      Ps.g_src = w.g_src; Ps.b_src = w.b_src;
      Ps.Wkp = pk.Wk; Ps.Wvp = pk.Wv;
      Ps.xs16 = xs16; Ps.ks16 = ks16; Ps.vs16 = vs16;
      Ps.do_src = 1; Ps.do_dst = 0; Ps.share = 0;
      int st = (nsrc + 15) / 16;
      attn_pre_kernel<<<(st + 3) / 4, 128, 0, stream>>>(Ps);
      PreArgs Pd{};
      Pd.x = xdst; Pd.n = N_;
      Pd.g_dst = w.g_dst; Pd.b_dst = w.b_dst;
      Pd.Wqp = pk.Wq; Pd.bq = w.bq;
      Pd.xd16 = xd16; Pd.q = qbuf;
      Pd.do_src = 0; Pd.do_dst = 1; Pd.share = 0;
      int dt = (N_ + 15) / 16;
      attn_pre_kernel<<<(dt + 3) / 4, 128, 0, stream>>>(Pd);
    }
    fillf_kernel<<<(N_ * 8 + 255) / 256, 256, 0, stream>>>(mbuf, -1e30f, (long)N_ * 8);
    fillf_kernel<<<(N_ * 8 + 255) / 256, 256, 0, stream>>>(denb, 0.f, (long)N_ * 8);
    fillf_kernel<<<(N_ * H_ + 255) / 256, 256, 0, stream>>>(aggb, 0.f, (long)N_ * H_);

    EdgeArgs Eg{};
    Eg.r16 = r16; Eg.E = E; Eg.g_r = w.g_r; Eg.b_r = w.b_r;
    Eg.Wkrp = pk.Wkr; Eg.Wvrp = pk.Wvr;
    Eg.ks16 = ks16; Eg.vs16 = vs16; Eg.q = qbuf;
    Eg.src = ei; Eg.dst = ei + E; Eg.sim = simb; Eg.vedge = vedge;
    int etiles = (E + 15) / 16;
    edge_kv_kernel<<<(etiles + 3) / 4, 128, 0, stream>>>(Eg);

    seg_max_kernel<<<(int)(((long)E * 8 + 255) / 256), 256, 0, stream>>>(simb, ei + E, E, mbuf);
    seg_expsum_kernel<<<(int)(((long)E * 8 + 255) / 256), 256, 0, stream>>>(simb, ei + E, E, mbuf, denb);
    seg_agg_kernel<<<(int)(((long)E * H_ + 255) / 256), 256, 0, stream>>>(simb, denb, vedge, ei + E, E, aggb);

    PostArgs Pp{};
    Pp.x = xdst; Pp.xd16 = bip ? xd16 : xs16; Pp.agg = aggb; Pp.n = N_;
    Pp.Wgp = pk.Wg; Pp.Wsp = pk.Ws; Pp.Wop = pk.Wo; Pp.W1fp = pk.W1f; Pp.W2fp = pk.W2f;
    Pp.bg = w.bg; Pp.bs = w.bs; Pp.bo = w.bo;
    Pp.g_ff = w.g_ff; Pp.b_ff = w.b_ff; Pp.b1f = w.b1f; Pp.b2f = w.b2f;
    int ptiles = (N_ + 15) / 16;
    attn_post_kernel<<<(ptiles + 1) / 2, 64, 0, stream>>>(Pp);
  };

  runAttn(topW, pk_top, x_t, x_t, N_, r16_patch, ei_patch, Ep, false);
  for (int l = 0; l < 2; ++l) {
    runAttn(tW[l], pk_t[l], x_t, x_t, N_, r16_t, ei_t, Et, false);
    runAttn(m2aW[l], pk_m[l], x_m, x_t, M_, r16_m2a, ei_m2a, Em, true);
    permute_kernel<<<(N_ * H_ + 255) / 256, 256, 0, stream>>>(x_t, x_perm, 1);
    runAttn(a2aW[l], pk_a[l], x_perm, x_perm, N_, r16_a2a, ei_a2a, Ea, false);
    permute_kernel<<<(N_ * H_ + 255) / 256, 256, 0, stream>>>(x_perm, x_t, 0);
  }
  copyf_kernel<<<(N_ * H_ + 255) / 256, 256, 0, stream>>>(x_t, (float*)d_out, (long)N_ * H_);
}
